// TurtleHeads_21869973471301
// MI455X (gfx1250) — matrix operand order repaired and checked by reading, not compiled, not run on hardware
//
#include <hip/hip_runtime.h>

typedef __bf16 bf16;
typedef __attribute__((ext_vector_type(16))) bf16  v16bf;
typedef __attribute__((ext_vector_type(8)))  bf16  v8bf;
typedef __attribute__((ext_vector_type(8)))  float v8f;

#define TH_LR   0.1f
#define TH_WD   1e-4f
#define TH_TEMP 0.7f
#define TH_B    4096
#define TH_K    128
#define TH_D    2688   // 384 + 768 + 1536
#define TH_PAD  40     // LDS row stride (elems): 16B-aligned, conflict-free b128 phases

__device__ __forceinline__ int head_dim(int h)  { return h == 0 ? 384 : (h == 1 ? 768 : 1536); }
__device__ __forceinline__ int head_dpre(int h) { return h == 0 ? 0   : (h == 1 ? 384 : 1152); }

union bfrag { v16bf v; v8bf h2[2]; };

// ---------------------------------------------------------------------------
// prep: L2-normalize feats -> bf16 row-major + bf16 transposed copies
// ---------------------------------------------------------------------------
__global__ __launch_bounds__(256) void th_prep_feats(
    const float* __restrict__ f0, const float* __restrict__ f1, const float* __restrict__ f2,
    bf16* __restrict__ fbf, bf16* __restrict__ ftr)
{
    const int row = blockIdx.x;
    const int h   = blockIdx.y;
    const int d   = head_dim(h);
    const float* src = (h == 0 ? f0 : (h == 1 ? f1 : f2)) + (size_t)row * d;

    __shared__ float red[8];
    float ss = 0.f;
    for (int i = threadIdx.x; i < d; i += 256) { float v = src[i]; ss += v * v; }
    #pragma unroll
    for (int m = 16; m >= 1; m >>= 1) ss += __shfl_xor(ss, m, 32);
    if ((threadIdx.x & 31) == 0) red[threadIdx.x >> 5] = ss;
    __syncthreads();
    float tot = 0.f;
    #pragma unroll
    for (int w = 0; w < 8; w++) tot += red[w];
    const float scale = 1.0f / fmaxf(sqrtf(tot), 1e-12f);

    bf16* dst  = fbf + (size_t)TH_B * head_dpre(h) + (size_t)row * d;
    bf16* dstT = ftr + (size_t)TH_B * head_dpre(h);
    for (int i = threadIdx.x; i < d; i += 256) {
        bf16 v = (bf16)(src[i] * scale);
        dst[i] = v;
        dstT[(size_t)i * TH_B + row] = v;
    }
}

// per-row sum of clipped targets (shared by all heads)
__global__ __launch_bounds__(128) void th_prep_srow(const float* __restrict__ tgt,
                                                    float* __restrict__ srow)
{
    const int row = blockIdx.x;
    float v = tgt[(size_t)row * TH_K + threadIdx.x];
    v = fminf(fmaxf(v, 1e-8f), 1.0f);
    #pragma unroll
    for (int m = 16; m >= 1; m >>= 1) v += __shfl_xor(v, m, 32);
    __shared__ float red[4];
    if ((threadIdx.x & 31) == 0) red[threadIdx.x >> 5] = v;
    __syncthreads();
    if (threadIdx.x == 0) srow[row] = red[0] + red[1] + red[2] + red[3];
}

// fp32 master copies of W/b + bf16 shadow of W
__global__ __launch_bounds__(256) void th_prep_params(
    const float* __restrict__ W0, const float* __restrict__ W1, const float* __restrict__ W2,
    const float* __restrict__ b0, const float* __restrict__ b1, const float* __restrict__ b2,
    float* __restrict__ Wm, bf16* __restrict__ Wbf, float* __restrict__ bm)
{
    const int h = blockIdx.y;
    const int d = head_dim(h);
    const size_t n   = (size_t)TH_K * d;
    const size_t idx = (size_t)blockIdx.x * 256 + threadIdx.x;
    const float* Ws = (h == 0 ? W0 : (h == 1 ? W1 : W2));
    const float* bs = (h == 0 ? b0 : (h == 1 ? b1 : b2));
    const size_t off = (size_t)TH_K * head_dpre(h);
    if (idx < n) {
        float w = Ws[idx];
        Wm[off + idx]  = w;
        Wbf[off + idx] = (bf16)w;
    }
    if (idx < TH_K) bm[h * TH_K + idx] = bs[idx];
}

// ---------------------------------------------------------------------------
// forward: Z = f Wt + b, logits = Z/TEMP, softmax, G = (s*p - t)/(B*TEMP)
// W k-chunk (128x32 bf16 = 8KB) staged in LDS, double-buffered, 1 barrier/chunk;
// B fragments ds_load_b128, software-pipelined so frag t+1 loads before WMMA t.
// mode 0: store G transposed (bf16 [128,4096]); mode 1: write f32 logits
// ---------------------------------------------------------------------------
__global__ __launch_bounds__(256) void th_fwd(
    const bf16* __restrict__ fbf, const bf16* __restrict__ Wbf,
    const float* __restrict__ bm, const float* __restrict__ tgt,
    const float* __restrict__ srow, bf16* __restrict__ Gt,
    float* __restrict__ out, int mode)
{
    const int lane = threadIdx.x & 31;
    const int wave = threadIdx.x >> 5;
    const int half = lane >> 4;
    const int l16  = lane & 15;
    const int h    = blockIdx.y;
    const int d    = head_dim(h);
    const bf16* f  = fbf + (size_t)TH_B * head_dpre(h);
    const bf16* W  = Wbf + (size_t)TH_K * head_dpre(h);
    const float* bv = bm + h * TH_K;
    const int rowbase = blockIdx.x * 128 + wave * 16;

    __shared__ bf16 sW[2][128 * TH_PAD];
    const int sr = threadIdx.x >> 1;          // 0..127: W row
    const int sq = threadIdx.x & 1;           // which 16-col half of the chunk
    const bf16* wsrc = W + (size_t)sr * d + sq * 16;
    bf16* sdst0 = &sW[0][sr * TH_PAD + sq * 16];
    bf16* sdst1 = &sW[1][sr * TH_PAD + sq * 16];

    { // stage chunk 0
        bfrag u;
        u.v = *(const v16bf*)(wsrc);
        *(v8bf*)(sdst0)     = u.h2[0];
        *(v8bf*)(sdst0 + 8) = u.h2[1];
    }
    __syncthreads();

    v8f acc[8] = {};
    const bf16* arow = f + (size_t)(rowbase + l16) * d;
    const int nck = d >> 5;

    for (int ci = 0; ci < nck; ci++) {
        const int k0 = ci << 5;
        const bool more = (ci + 1 < nck);

        bfrag nx;                                   // prefetch next W chunk (global)
        if (more) nx.v = *(const v16bf*)(wsrc + (size_t)(k0 + 32));

        // A fragment (16x32 bf16): two 16B chunks per lane, k = 16*half + s (contiguous)
        bfrag ua;
        ua.h2[0] = *(const v8bf*)(arow + k0 + half * 16);
        ua.h2[1] = *(const v8bf*)(arow + k0 + half * 16 + 8);
        const v16bf A = ua.v;

        // B fragments (32x16) from LDS: col = out-k (lane%16), K-half by lane/16.
        // Register-pipelined: fragment t+1 is loaded before WMMA t issues.
        const bf16* sb = &sW[ci & 1][half * 16];
        bfrag cur;
        cur.h2[0] = *(const v8bf*)(sb + l16 * TH_PAD);
        cur.h2[1] = *(const v8bf*)(sb + l16 * TH_PAD + 8);
        #pragma unroll
        for (int t = 0; t < 8; t++) {
            bfrag nb = cur;
            if (t < 7) {
                const bf16* pnb = sb + ((t + 1) * 16 + l16) * TH_PAD;
                nb.h2[0] = *(const v8bf*)(pnb);
                nb.h2[1] = *(const v8bf*)(pnb + 8);
            }
            acc[t] = __builtin_amdgcn_wmma_f32_16x16x32_bf16(
                false, A, false, cur.v, (short)0, acc[t], false, false);
            cur = nb;
        }

        if (more) {
            bf16* sd = (ci & 1) ? sdst0 : sdst1;
            *(v8bf*)(sd)     = nx.h2[0];
            *(v8bf*)(sd + 8) = nx.h2[1];
            __syncthreads();
        }
    }

    const float invT = 1.0f / TH_TEMP;
    float bcol[8];
    #pragma unroll
    for (int t = 0; t < 8; t++) bcol[t] = bv[t * 16 + l16];
    #pragma unroll
    for (int t = 0; t < 8; t++)
        #pragma unroll
        for (int r = 0; r < 8; r++)
            acc[t][r] = (acc[t][r] + bcol[t]) * invT;

    // softmax per row; halves (lanes 0-15 / 16-31) are distinct rows
    #pragma unroll
    for (int r = 0; r < 8; r++) {
        float mx = -3.4e38f;
        #pragma unroll
        for (int t = 0; t < 8; t++) mx = fmaxf(mx, acc[t][r]);
        #pragma unroll
        for (int m = 8; m >= 1; m >>= 1) mx = fmaxf(mx, __shfl_xor(mx, m, 32));
        float e[8], se = 0.f;
        #pragma unroll
        for (int t = 0; t < 8; t++) { e[t] = __expf(acc[t][r] - mx); se += e[t]; }
        #pragma unroll
        for (int m = 8; m >= 1; m >>= 1) se += __shfl_xor(se, m, 32);
        const int row = rowbase + r + 8 * half;
        if (mode == 0) {
            const float sr2 = srow[row];
            const float gs  = 1.0f / ((float)TH_B * TH_TEMP);
            const float inv = 1.0f / se;
            #pragma unroll
            for (int t = 0; t < 8; t++) {
                float tv = tgt[(size_t)row * TH_K + t * 16 + l16];
                tv = fminf(fmaxf(tv, 1e-8f), 1.0f);
                acc[t][r] = (sr2 * e[t] * inv - tv) * gs;
            }
        } else {
            #pragma unroll
            for (int t = 0; t < 8; t++)
                out[(size_t)h * TH_B * TH_K + (size_t)row * TH_K + t * 16 + l16] = acc[t][r];
        }
    }

    if (mode == 0) {
        bf16* G = Gt + (size_t)h * TH_K * TH_B;
        #pragma unroll
        for (int t = 0; t < 8; t++) {
            v8bf pk;
            #pragma unroll
            for (int r = 0; r < 8; r++) pk[r] = (bf16)acc[t][r];
            *(v8bf*)(G + (size_t)(t * 16 + l16) * TH_B + rowbase + 8 * half) = pk;
        }
    }
}

// ---------------------------------------------------------------------------
// backward + update: gW = Gt x f (reduce over batch 4096), gb from A-frags,
// fp32 SGD with weight decay on master W/b, refresh bf16 W shadow.
// F batch-chunk (64 cols x 32 batch = 4KB) staged in LDS, double-buffered;
// B fragments register-pipelined like the forward kernel.
// WG = 128 krows x 64 dcols; wave = 16 krows x 64 dcols (4 WMMA tiles)
// ---------------------------------------------------------------------------
__global__ __launch_bounds__(256) void th_bwd(
    const bf16* __restrict__ Gt, const bf16* __restrict__ ftr,
    float* __restrict__ Wm, bf16* __restrict__ Wbf, float* __restrict__ bm)
{
    const int lane = threadIdx.x & 31;
    const int wave = threadIdx.x >> 5;
    const int half = lane >> 4;
    const int l16  = lane & 15;
    const int h    = blockIdx.y;
    const int d    = head_dim(h);
    const int j0   = blockIdx.x * 64;
    if (j0 >= d) return;

    const bf16* G = Gt  + (size_t)h * TH_K * TH_B;
    const bf16* F = ftr + (size_t)TH_B * head_dpre(h);
    const int k0  = wave * 16;

    __shared__ bf16 sF[2][64 * TH_PAD];
    const int sc = threadIdx.x >> 2;          // 0..63: relative feature column
    const int sq = threadIdx.x & 3;           // 8-elem batch quarter of the chunk
    const bf16* fsrc = F + (size_t)(j0 + sc) * TH_B + sq * 8;
    bf16* fdst0 = &sF[0][sc * TH_PAD + sq * 8];
    bf16* fdst1 = &sF[1][sc * TH_PAD + sq * 8];

    *(v8bf*)fdst0 = *(const v8bf*)(fsrc);     // stage chunk 0 (bs = 0)
    __syncthreads();

    v8f acc[4] = {};
    float gb = 0.f;
    const bf16* grow = G + (size_t)(k0 + l16) * TH_B;

    for (int ci = 0; ci < (TH_B >> 5); ci++) {
        const int bs = ci << 5;
        const bool more = (ci + 1 < (TH_B >> 5));

        v8bf nx;                               // prefetch next F chunk (global)
        if (more) nx = *(const v8bf*)(fsrc + (size_t)(bs + 32));

        bfrag ua;                              // A fragment from Gt (global)
        ua.h2[0] = *(const v8bf*)(grow + bs + half * 16);
        ua.h2[1] = *(const v8bf*)(grow + bs + half * 16 + 8);
        const v16bf A = ua.v;
        if (blockIdx.x == 0) {
            #pragma unroll
            for (int i = 0; i < 8; i++) gb += (float)ua.h2[0][i] + (float)ua.h2[1][i];
        }

        const bf16* sb = &sF[ci & 1][half * 16];
        bfrag cur;
        cur.h2[0] = *(const v8bf*)(sb + l16 * TH_PAD);
        cur.h2[1] = *(const v8bf*)(sb + l16 * TH_PAD + 8);
        #pragma unroll
        for (int jt = 0; jt < 4; jt++) {
            bfrag nb = cur;
            if (jt < 3) {
                const bf16* pnb = sb + ((jt + 1) * 16 + l16) * TH_PAD;
                nb.h2[0] = *(const v8bf*)(pnb);
                nb.h2[1] = *(const v8bf*)(pnb + 8);
            }
            acc[jt] = __builtin_amdgcn_wmma_f32_16x16x32_bf16(
                false, A, false, cur.v, (short)0, acc[jt], false, false);
            cur = nb;
        }

        if (more) {
            *(v8bf*)((ci & 1) ? fdst0 : fdst1) = nx;
            __syncthreads();
        }
    }

    float* Wmh = Wm  + (size_t)TH_K * head_dpre(h);
    bf16*  Wbh = Wbf + (size_t)TH_K * head_dpre(h);
    #pragma unroll
    for (int jt = 0; jt < 4; jt++) {
        #pragma unroll
        for (int r = 0; r < 8; r++) {
            const int krow = k0 + r + 8 * half;
            const int col  = j0 + jt * 16 + l16;
            const size_t idx = (size_t)krow * d + col;
            float w = Wmh[idx];
            w -= TH_LR * (acc[jt][r] + TH_WD * w);
            Wmh[idx] = w;
            Wbh[idx] = (bf16)w;
        }
    }

    if (blockIdx.x == 0) {
        gb += __shfl_xor(gb, 16, 32);          // lane pair covers all 32 batch elems
        if (lane < 16) {
            float* bp = bm + h * TH_K + k0 + lane;
            float bb = *bp;
            bb -= TH_LR * (gb + TH_WD * bb);
            *bp = bb;
        }
    }
}

// ---------------------------------------------------------------------------
extern "C" void kernel_launch(void* const* d_in, const int* in_sizes, int n_in,
                              void* d_out, int out_size, void* d_ws, size_t ws_size,
                              hipStream_t stream)
{
    (void)in_sizes; (void)n_in; (void)out_size; (void)ws_size;
    const float* f0 = (const float*)d_in[0];
    const float* W0 = (const float*)d_in[1];
    const float* b0 = (const float*)d_in[2];
    const float* f1 = (const float*)d_in[3];
    const float* W1 = (const float*)d_in[4];
    const float* b1 = (const float*)d_in[5];
    const float* f2 = (const float*)d_in[6];
    const float* W2 = (const float*)d_in[7];
    const float* b2 = (const float*)d_in[8];
    const float* tgt = (const float*)d_in[9];
    float* out = (float*)d_out;

    char* p = (char*)d_ws;
    auto take = [&](size_t bytes) -> char* {
        char* q = p;
        p += (bytes + 255) & ~(size_t)255;
        return q;
    };
    bf16*  fbf  = (bf16*)take((size_t)TH_B * TH_D * 2);     // normalized feats, row-major
    bf16*  ftr  = (bf16*)take((size_t)TH_B * TH_D * 2);     // normalized feats, transposed
    float* Wm   = (float*)take((size_t)TH_K * TH_D * 4);    // fp32 master W
    bf16*  Wbf  = (bf16*)take((size_t)TH_K * TH_D * 2);     // bf16 shadow W
    float* bm   = (float*)take((size_t)3 * TH_K * 4);       // fp32 master b
    bf16*  Gt   = (bf16*)take((size_t)3 * TH_K * TH_B * 2); // G transposed per head
    float* srow = (float*)take((size_t)TH_B * 4);           // sum of clipped targets

    th_prep_feats<<<dim3(TH_B, 3), 256, 0, stream>>>(f0, f1, f2, fbf, ftr);
    th_prep_srow<<<TH_B, 128, 0, stream>>>(tgt, srow);
    th_prep_params<<<dim3(768, 3), 256, 0, stream>>>(W0, W1, W2, b0, b1, b2, Wm, Wbf, bm);

    for (int m = 0; m < 100; m++) {
        th_fwd<<<dim3(32, 3), 256, 0, stream>>>(fbf, Wbf, bm, tgt, srow, Gt, nullptr, 0);
        th_bwd<<<dim3(24, 3), 256, 0, stream>>>(Gt, ftr, Wm, Wbf, bm);
    }
    th_fwd<<<dim3(32, 3), 256, 0, stream>>>(fbf, Wbf, bm, tgt, srow, Gt, out, 1);
}
